// KNNAttention_39213051413023
// MI455X (gfx1250) — matrix operand order repaired and checked by reading, not compiled, not run on hardware
//
#include <hip/hip_runtime.h>
#include <hip/hip_bf16.h>

typedef __attribute__((ext_vector_type(16))) _Float16 v16h;
typedef __attribute__((ext_vector_type(8)))  _Float16 v8h;
typedef __attribute__((ext_vector_type(8)))  float    v8f;
typedef __attribute__((ext_vector_type(4)))  float    f32x4;

#define B_ 2
#define H_ 8
#define S_ 2048
#define D_ 64
#define K_ 32
#define NEGMAX_ 3.402823466e38f

// ---------------------------------------------------------------------------
// Pass 1a: L2-normalize rows of length D and emit f16 into workspace.
// ---------------------------------------------------------------------------
__global__ void qk_norm_f16(const float* __restrict__ src,
                            _Float16* __restrict__ dst, int nrows) {
    int row = blockIdx.x * blockDim.x + threadIdx.x;
    if (row >= nrows) return;
    const float* p = src + (size_t)row * D_;
    float ss = 0.f;
#pragma unroll
    for (int d = 0; d < D_; ++d) { float x = p[d]; ss += x * x; }
    float inv = 1.0f / fmaxf(sqrtf(ss), 1e-12f);
    _Float16* o = dst + (size_t)row * D_;
#pragma unroll
    for (int d = 0; d < D_; ++d) o[d] = (_Float16)(p[d] * inv);
}

// ---------------------------------------------------------------------------
// Pass 1b: V f32 [b][key][d] -> f16 transposed [b][d][key], so the PV
// B-operand is a single contiguous 32B (v16h) load per WMMA.
// ---------------------------------------------------------------------------
__global__ void v_transpose_f16(const float* __restrict__ src,
                                _Float16* __restrict__ dst, int n) {
    int i = blockIdx.x * blockDim.x + threadIdx.x;   // i = ((b*D + d)*S + key)
    if (i >= n) return;
    int key = i % S_;
    int d   = (i / S_) % D_;
    int b   = i / (S_ * D_);
    dst[i] = (_Float16)src[((size_t)b * S_ + key) * D_ + d];
}

// ---------------------------------------------------------------------------
// Pass 2: flash attention with KNN-memory branch.
// 128 threads = 4 waves; wave w handles 16 query rows of one (b,h).
// grid = (S/64, B*H)
// ---------------------------------------------------------------------------
__launch_bounds__(128, 1)
__global__ void knn_attn_fwd(const _Float16* __restrict__ qn,
                             const _Float16* __restrict__ kn,
                             const _Float16* __restrict__ vt,
                             const float* __restrict__ maskg,
                             const float* __restrict__ mem_k,
                             const float* __restrict__ mem_v,
                             const unsigned char* __restrict__ mem_mask,
                             const float* __restrict__ rpb,
                             const float* __restrict__ scale_log,
                             float* __restrict__ out) {
    const int lane = threadIdx.x & 31;
    const int wave = threadIdx.x >> 5;
    const int l15  = lane & 15;
    const int g    = lane >> 4;           // which 16-lane half

    const int bh    = blockIdx.y;
    const int b     = bh / H_;
    const int h     = bh % H_;
    const int qbase = blockIdx.x * 64 + wave * 16;

    const float sc = __expf(scale_log[h]);

    __shared__ __align__(16) _Float16 qn_lds[4][16 * D_];   // per-wave Q tile
    __shared__ __align__(16) float    pmem_lds[4][16 * K_]; // mem-branch probs
    __shared__ __align__(16) _Float16 pl_lds[4][16 * 32];   // P tile (C->A)

    _Float16* qtile = qn_lds[wave];
    float*    pmem  = pmem_lds[wave];
    _Float16* pl    = pl_lds[wave];

    // ---- stage Q tile (16x64 f16 = 2KB) into LDS with 16B copies ----
    {
        const uint4* src4 = (const uint4*)(qn + ((size_t)bh * S_ + qbase) * D_);
        uint4* dst4 = (uint4*)qtile;
#pragma unroll
        for (int t = 0; t < 4; ++t) dst4[t * 32 + lane] = src4[t * 32 + lane];
    }

    // ---- build A operands (16x32 f16, two D-chunks): 2x 16B LDS loads ----
    v16h aq0, aq1;
#pragma unroll
    for (int dc = 0; dc < 2; ++dc) {
        v8h lo = *(const v8h*)(qtile + l15 * D_ + dc * 32 + g * 16);
        v8h hi = *(const v8h*)(qtile + l15 * D_ + dc * 32 + g * 16 + 8);
        v16h a;
#pragma unroll
        for (int e = 0; e < 8; ++e) { a[e] = lo[e]; a[8 + e] = hi[e]; }
        if (dc == 0) aq0 = a; else aq1 = a;
    }

    v8f acc0 = {}, acc1 = {}, acc2 = {}, acc3 = {};
    float mrow[8], lrow[8], m0row[8];

    // ---- memory branch: sim_mem = q . mem_k[i] (per-row private keys) ----
    {
        const int key = lane;  // one slot per lane
        const unsigned char* mm = mem_mask + ((size_t)bh * S_ + qbase) * K_ + key;
        const float* mk = mem_k + (((size_t)bh * S_ + qbase) * K_ + key) * (size_t)D_;
        for (int t = 0; t < 16; ++t) {
            const f32x4* mkr = (const f32x4*)(mk + (size_t)t * K_ * D_);
            __builtin_prefetch(mk + (size_t)(t + 1) * K_ * D_, 0, 1);
            float dot = 0.f;
#pragma unroll
            for (int d4 = 0; d4 < D_ / 4; ++d4) {
                f32x4 m4 = __builtin_nontemporal_load(&mkr[d4]);
                dot += (float)qtile[t * D_ + 4 * d4 + 0] * m4[0];
                dot += (float)qtile[t * D_ + 4 * d4 + 1] * m4[1];
                dot += (float)qtile[t * D_ + 4 * d4 + 2] * m4[2];
                dot += (float)qtile[t * D_ + 4 * d4 + 3] * m4[3];
            }
            float s = dot * sc;
            if (!mm[(size_t)t * K_]) s = -NEGMAX_;
            float m = s;
            for (int o = 16; o >= 1; o >>= 1) m = fmaxf(m, __shfl_xor(m, o, 32));
            float p = __expf(s - m);
            float l = p;
            for (int o = 16; o >= 1; o >>= 1) l += __shfl_xor(l, o, 32);
            pmem[t * K_ + key] = p;
            if ((t >> 3) == g) { mrow[t & 7] = m; lrow[t & 7] = l; m0row[t & 7] = m; }
        }
    }

    const _Float16* knb   = kn  + (size_t)b * S_ * D_;
    const _Float16* vtb   = vt  + (size_t)b * D_ * S_;
    const float*    maskb = maskg + (size_t)b * S_;
    const float*    rpbh  = rpb + (size_t)h * S_ * S_;

    // ---- causal flash loop over 32-key chunks ----
    for (int kb = 0; kb < qbase + 16; kb += 32) {
        // S = Q Kt : two 16-key N tiles, k=64 via two chained WMMAs each
        v8f s[2];
#pragma unroll
        for (int jt = 0; jt < 2; ++jt) {
            const _Float16* kr = knb + (size_t)(kb + jt * 16 + l15) * D_;
            v16h b0 = *(const v16h*)(kr + g * 16);        // d = 0..31 chunk
            v16h b1 = *(const v16h*)(kr + 32 + g * 16);   // d = 32..63 chunk
            v8f c = {};
            c = __builtin_amdgcn_wmma_f32_16x16x32_f16(false, aq0, false, b0,
                                                       (short)0, c, false, false);
            c = __builtin_amdgcn_wmma_f32_16x16x32_f16(false, aq1, false, b1,
                                                       (short)0, c, false, false);
            s[jt] = c;
        }

        // key-padding penalty is row-invariant: hoist out of the r loop
        const float pen0 = -NEGMAX_ * (1.0f - maskb[kb + l15]);
        const float pen1 = -NEGMAX_ * (1.0f - maskb[kb + 16 + l15]);

        // scale + rel_pos_bias + masks + online softmax update (C layout)
        float p0[8], p1[8];
#pragma unroll
        for (int r = 0; r < 8; ++r) {
            const int grow = qbase + r + 8 * g;
            const float* rpbrow = rpbh + (size_t)grow * S_ + kb;
            float sv0, sv1;
            {
                int j = kb + l15;
                float val = s[0][r] * sc + __builtin_nontemporal_load(&rpbrow[l15]);
                val += pen0;
                if (j > grow) val = -NEGMAX_;
                sv0 = val;
            }
            {
                int j = kb + 16 + l15;
                float val = s[1][r] * sc + __builtin_nontemporal_load(&rpbrow[16 + l15]);
                val += pen1;
                if (j > grow) val = -NEGMAX_;
                sv1 = val;
            }
            float m_ = fmaxf(sv0, sv1);
            for (int o = 8; o >= 1; o >>= 1) m_ = fmaxf(m_, __shfl_xor(m_, o, 32));
            m_ = fmaxf(m_, mrow[r]);
            float scf = __expf(mrow[r] - m_);
            p0[r] = __expf(sv0 - m_);
            p1[r] = __expf(sv1 - m_);
            float rsum = p0[r] + p1[r];
            for (int o = 8; o >= 1; o >>= 1) rsum += __shfl_xor(rsum, o, 32);
            lrow[r] = lrow[r] * scf + rsum;
            mrow[r] = m_;
            acc0[r] *= scf; acc1[r] *= scf; acc2[r] *= scf; acc3[r] *= scf;
            pl[(r + 8 * g) * 32 + l15]      = (_Float16)p0[r];
            pl[(r + 8 * g) * 32 + 16 + l15] = (_Float16)p1[r];
        }

        // P: C-layout -> A-layout through LDS, two 16B ds loads per lane
        v16h pa;
        {
            v8h lo = *(const v8h*)(pl + l15 * 32 + g * 16);
            v8h hi = *(const v8h*)(pl + l15 * 32 + g * 16 + 8);
#pragma unroll
            for (int e = 0; e < 8; ++e) { pa[e] = lo[e]; pa[8 + e] = hi[e]; }
        }

        // O += P V : 4 D-column tiles, k=32 keys each.
        // V is pre-transposed f16 [d][key] -> each B operand is one 32B load.
#pragma unroll
        for (int jt = 0; jt < 4; ++jt) {
            v16h bv = *(const v16h*)(vtb + (size_t)(jt * 16 + l15) * S_ + kb + g * 16);
            if (jt == 0)
                acc0 = __builtin_amdgcn_wmma_f32_16x16x32_f16(false, pa, false, bv,
                                                              (short)0, acc0, false, false);
            else if (jt == 1)
                acc1 = __builtin_amdgcn_wmma_f32_16x16x32_f16(false, pa, false, bv,
                                                              (short)0, acc1, false, false);
            else if (jt == 2)
                acc2 = __builtin_amdgcn_wmma_f32_16x16x32_f16(false, pa, false, bv,
                                                              (short)0, acc2, false, false);
            else
                acc3 = __builtin_amdgcn_wmma_f32_16x16x32_f16(false, pa, false, bv,
                                                              (short)0, acc3, false, false);
        }
    }

    // ---- fold memory branch back in and normalize ----
#pragma unroll
    for (int r = 0; r < 8; ++r) {
        const int grow = qbase + r + 8 * g;
        const float fm = __expf(m0row[r] - mrow[r]);
        const float* mvrow = mem_v + (((size_t)bh * S_ + grow) * K_) * (size_t)D_;
        __builtin_prefetch(mvrow, 0, 1);
        float m_acc[4] = {0.f, 0.f, 0.f, 0.f};
        for (int kk = 0; kk < K_; ++kk) {
            float p = pmem[(r + 8 * g) * K_ + kk];
            const float* mv = mvrow + (size_t)kk * D_;
#pragma unroll
            for (int jt = 0; jt < 4; ++jt)
                m_acc[jt] += p * __builtin_nontemporal_load(&mv[jt * 16 + l15]);
        }
        float inv_l = 1.0f / lrow[r];
        float* op = out + ((size_t)bh * S_ + grow) * D_;
        op[0 * 16 + l15] = (acc0[r] + fm * m_acc[0]) * inv_l;
        op[1 * 16 + l15] = (acc1[r] + fm * m_acc[1]) * inv_l;
        op[2 * 16 + l15] = (acc2[r] + fm * m_acc[2]) * inv_l;
        op[3 * 16 + l15] = (acc3[r] + fm * m_acc[3]) * inv_l;
    }
}

// ---------------------------------------------------------------------------
extern "C" void kernel_launch(void* const* d_in, const int* in_sizes, int n_in,
                              void* d_out, int out_size, void* d_ws, size_t ws_size,
                              hipStream_t stream) {
    const float* q        = (const float*)d_in[0];
    const float* k        = (const float*)d_in[1];
    const float* v        = (const float*)d_in[2];
    const float* mask     = (const float*)d_in[3];
    const float* mem_k    = (const float*)d_in[4];
    const float* mem_v    = (const float*)d_in[5];
    const unsigned char* mem_mask = (const unsigned char*)d_in[6];
    const float* rpb      = (const float*)d_in[7];
    const float* scale    = (const float*)d_in[8];
    float* out            = (float*)d_out;

    _Float16* qn = (_Float16*)d_ws;                      // 4 MB
    _Float16* kn = qn + (size_t)B_ * H_ * S_ * D_;       // +0.5 MB
    _Float16* vt = kn + (size_t)B_ * S_ * D_;            // +0.5 MB (transposed)

    const int qrows = B_ * H_ * S_;
    const int krows = B_ * S_;
    const int vn    = B_ * S_ * D_;
    qk_norm_f16<<<(qrows + 255) / 256, 256, 0, stream>>>(q, qn, qrows);
    qk_norm_f16<<<(krows + 255) / 256, 256, 0, stream>>>(k, kn, krows);
    v_transpose_f16<<<(vn + 255) / 256, 256, 0, stream>>>(v, vt, vn);

    dim3 grid(S_ / 64, B_ * H_);
    knn_attn_fwd<<<grid, 128, 0, stream>>>(qn, kn, vt, mask, mem_k, mem_v,
                                           mem_mask, rpb, scale, out);
}
